// ProteinLigandTNP_36661840839006
// MI455X (gfx1250) — compile-verified, matrix operand order checked by reading, not run on hardware
//
#include <hip/hip_runtime.h>
#include <hip/hip_bf16.h>
#include <math.h>

// ---------------------------------------------------------------------------
// Problem constants (from the reference)
// ---------------------------------------------------------------------------
constexpr int B_    = 4;
constexpr int N_    = 1024;   // NCTX + NQRY
constexpr int NCTX_ = 768;
constexpr int D_    = 256;
constexpr int H_    = 8;
constexpr int HD_   = 32;     // head dim == one WMMA K step (16x16x32 f16)
constexpr int L_    = 4;
constexpr int M_    = B_ * N_;   // 4096 rows

typedef __attribute__((ext_vector_type(16))) _Float16 v16h;
typedef __attribute__((ext_vector_type(8)))  _Float16 v8h;
typedef __attribute__((ext_vector_type(8)))  float    v8f;

union AFrag { v16h v; v8h h[2]; };

// WMMA f16 A-matrix 16x32 layout (CDNA5 ISA 7.12.2):
//   lane<16  : row M=lane,    K = {0..7, 16..23}   (two 16B chunks)
//   lane>=16 : row M=lane-16, K = {8..15, 24..31}
// B-matrix (32x16) mirrors this with lane&15 indexing the N column.
__device__ inline AFrag load_frag_rowmajor(const _Float16* base, int row_stride_h,
                                           int row, int kchunk_base, int lane) {
    AFrag f;
    int kb = (lane < 16) ? 0 : 8;
    const _Float16* p = base + (size_t)row * row_stride_h + kchunk_base;
    f.h[0] = *(const v8h*)(p + kb);
    f.h[1] = *(const v8h*)(p + kb + 16);
    return f;
}

// ---------------------------------------------------------------------------
// CDNA5 async global->LDS copy (ASYNCcnt path, no VGPR landing zone).
// GVS addressing: SGPR 64-bit base + per-lane 32-bit byte offset.
// The LDS destination address is the low 32 bits of the generic pointer
// (shared-aperture flat addresses carry the wave-relative LDS offset there).
// Async loads complete IN ORDER, so a partial s_wait_asynccnt <= N is enough
// to guarantee the older tile has landed while the next one is still in
// flight -> true double buffering through the async engine.
// ---------------------------------------------------------------------------
__device__ inline unsigned lds_off_of(const void* p) {
    return (unsigned)(unsigned long long)(uintptr_t)p;
}

__device__ inline void async_copy_b128(unsigned lds_byte_off, const void* gbase,
                                       unsigned gbyte_off) {
    asm volatile("global_load_async_to_lds_b128 %0, %1, %2 offset:0"
                 :: "v"(lds_byte_off), "v"(gbyte_off),
                    "s"((unsigned long long)(uintptr_t)gbase)
                 : "memory");
}

template <int Nw>
__device__ inline void wait_async() {
#if __has_builtin(__builtin_amdgcn_s_wait_asynccnt)
    __builtin_amdgcn_s_wait_asynccnt(Nw);
#else
    asm volatile("s_wait_asynccnt %0" :: "i"(Nw) : "memory");
#endif
}

// ---------------------------------------------------------------------------
// fp32 -> fp16 weight conversion (once per launch; weights then live in L2)
// ---------------------------------------------------------------------------
__global__ void f32_to_f16_kernel(const float* __restrict__ in,
                                  _Float16* __restrict__ out, int n) {
    int i = blockIdx.x * blockDim.x + threadIdx.x;
    if (i < n) out[i] = (_Float16)in[i];
}

// ---------------------------------------------------------------------------
// LayerNorm over D=256, one wave32 per row, output f16 for WMMA consumption
// ---------------------------------------------------------------------------
__global__ __launch_bounds__(256)
void ln_kernel(const float* __restrict__ x, const float* __restrict__ g,
               const float* __restrict__ be, _Float16* __restrict__ h) {
    int row  = blockIdx.x * 8 + (threadIdx.x >> 5);
    int lane = threadIdx.x & 31;
    const float* xr = x + (size_t)row * D_;

    float v[8];
    float s = 0.f;
#pragma unroll
    for (int i = 0; i < 8; i++) { v[i] = xr[i * 32 + lane]; s += v[i]; }
#pragma unroll
    for (int off = 16; off >= 1; off >>= 1) s += __shfl_xor(s, off, 32);
    float mean = s * (1.f / 256.f);

    float var = 0.f;
#pragma unroll
    for (int i = 0; i < 8; i++) { float d = v[i] - mean; var += d * d; }
#pragma unroll
    for (int off = 16; off >= 1; off >>= 1) var += __shfl_xor(var, off, 32);
    float rstd = rsqrtf(var * (1.f / 256.f) + 1e-5f);

#pragma unroll
    for (int i = 0; i < 8; i++) {
        int c = i * 32 + lane;
        h[(size_t)row * D_ + c] = (_Float16)((v[i] - mean) * rstd * g[c] + be[c]);
    }
}

// ---------------------------------------------------------------------------
// Tiled WMMA GEMM:  out[M, NOUT] = A[M, KDIM] (f16) @ W[NOUT, KDIM]^T (f16)
// Block = 4 waves -> 64(M) x 128(N) tile; each wave: one A fragment feeds
// 8 WMMAs per 32-wide K step.  Double-buffered async-LDS staging:
//   per tile 6 async b128 ops/thread; steady-state wait = s_wait_asynccnt 6.
// EPI: 0 = +bias, store f16
//      1 = gelu(+bias), store f16
//      2 = +bias + residual(f32), store f32 (in-place x update)
// ---------------------------------------------------------------------------
template <int KDIM, int EPI>
__global__ __launch_bounds__(128)
void gemm_wmma_kernel(const _Float16* __restrict__ A,
                      const _Float16* __restrict__ W,
                      const float* __restrict__ bias,
                      const float* __restrict__ resid,
                      _Float16* __restrict__ out16,
                      float* __restrict__ out32, int NOUT) {
    constexpr int NK = KDIM / 32;
    __shared__ __align__(16) _Float16 As[2][64][32];
    __shared__ __align__(16) _Float16 Bs[2][128][32];

    int mblk = blockIdx.x, nblk = blockIdx.y;
    int tid = threadIdx.x, wave = tid >> 5, lane = tid & 31;
    int am = lane & 15, mloc = 8 * (lane >> 4);

    auto issue_tile = [&](int buf, int kt) {
        int k0 = kt * 32;
        unsigned aB = lds_off_of(&As[buf][0][0]);
        unsigned bB = lds_off_of(&Bs[buf][0][0]);
#pragma unroll
        for (int j = 0; j < 2; j++) {          // 64x32 A tile: 256 b128 chunks
            int c = tid + j * 128;
            int r = c >> 2, colh = (c & 3) * 8;
            async_copy_b128(aB + c * 16, A,
                            (unsigned)(((mblk * 64 + r) * KDIM + k0 + colh) * 2));
        }
#pragma unroll
        for (int j = 0; j < 4; j++) {          // 128x32 B tile: 512 b128 chunks
            int c = tid + j * 128;
            int r = c >> 2, colh = (c & 3) * 8;
            async_copy_b128(bB + c * 16, W,
                            (unsigned)(((nblk * 128 + r) * KDIM + k0 + colh) * 2));
        }
    };

    v8f acc[8];
#pragma unroll
    for (int t = 0; t < 8; t++) acc[t] = (v8f){0.f,0.f,0.f,0.f,0.f,0.f,0.f,0.f};

    issue_tile(0, 0);
#pragma unroll
    for (int kt = 0; kt < NK; kt++) {
        int cur = kt & 1;
        if (kt + 1 < NK) {                      // stream next tile while we wait
            issue_tile(cur ^ 1, kt + 1);
            wait_async<6>();                    // partial wait: current tile done
        } else {
            wait_async<0>();
        }
        __syncthreads();

        AFrag a = load_frag_rowmajor(&As[cur][0][0], 32, wave * 16 + am, 0, lane);
#pragma unroll
        for (int t = 0; t < 8; t++) {
            AFrag b = load_frag_rowmajor(&Bs[cur][0][0], 32, t * 16 + am, 0, lane);
            acc[t] = __builtin_amdgcn_wmma_f32_16x16x32_f16(
                false, a.v, false, b.v, (short)0, acc[t], false, false);
        }
        __syncthreads();                        // release 'cur' for overwrite
    }

#pragma unroll
    for (int t = 0; t < 8; t++) {
        int n = nblk * 128 + t * 16 + (lane & 15);
        float bv = bias[n];
#pragma unroll
        for (int i = 0; i < 8; i++) {
            int m = mblk * 64 + wave * 16 + mloc + i;   // C layout: M = vgpr + 8*(lane>=16)
            float v = acc[t][i] + bv;
            if constexpr (EPI == 1)
                v = 0.5f * v * (1.f + erff(v * 0.70710678118654752f));  // exact gelu
            if constexpr (EPI == 2) {
                size_t idx = (size_t)m * NOUT + n;
                out32[idx] = v + resid[idx];
            } else {
                out16[(size_t)m * NOUT + n] = (_Float16)v;
            }
        }
    }
}

// ---------------------------------------------------------------------------
// Flash-style attention. grid = (B*H, N/64). Block = 4 waves; each wave owns
// 16 query rows. Streams 64-key blocks with a double-buffered async K tile:
//   K tiles   : async global->LDS b128 (next tile overlaps V staging+compute)
//   scores    : 4x  v_wmma_f32_16x16x32_f16 (hd=32 == one K step)
//   softmax   : online (row stats via shfl within 16-lane halves)
//   P@V       : P -> LDS (C->A layout fix-up), V staged transposed; 4x WMMA
// Bias = -alpha*log(ppr) on ctx keys; trust gate folded into V staging.
// Mask: key>=NCTX is only visible to query==key (self), per reference mask.
// ---------------------------------------------------------------------------
__global__ __launch_bounds__(128)
void attn_kernel(const _Float16* __restrict__ qkv,   // [B, N, 3*D] f16
                 const float* __restrict__ ctx_ppr,  // [B, NCTX]
                 const float* __restrict__ ctx_trust,// [B, NCTX]
                 const float* __restrict__ alpha_p,
                 const float* __restrict__ tscale_p, int layer,
                 _Float16* __restrict__ o)            // [B, N, D] f16
{
    __shared__ __align__(16) _Float16 Qs[64][32];
    __shared__ __align__(16) _Float16 Ks[2][64][32];  // double-buffered
    __shared__ __align__(16) _Float16 Vt[32][64];     // transposed (hd-major)
    __shared__ __align__(16) _Float16 Ps[4][16][64];  // per-wave P tile
    __shared__ float cbias[64];

    int bh = blockIdx.x;
    int b = bh >> 3, hh = bh & 7;
    int qt = blockIdx.y;
    int tid = threadIdx.x, wave = tid >> 5, lane = tid & 31;
    int am = lane & 15, mloc = 8 * (lane >> 4);

    float alpha = alpha_p[layer];
    float tsc   = tscale_p[layer];
    const float scale = 0.17677669529663687f;  // 1/sqrt(32)

    auto issue_ktile = [&](int buf, int kb0) {
        unsigned kB = lds_off_of(&Ks[buf][0][0]);
#pragma unroll
        for (int j = 0; j < 2; j++) {
            int c = tid + j * 128;
            int r = c >> 2, colh = (c & 3) * 8;
            async_copy_b128(kB + c * 16, qkv,
                (unsigned)((((b * N_ + kb0 + r) * (3 * D_)) + D_ + hh * HD_ + colh) * 2));
        }
    };

    // stage Q tile (64 queries x 32 hd) + first K tile via async copies
    {
        unsigned qsBase = lds_off_of(&Qs[0][0]);
#pragma unroll
        for (int j = 0; j < 2; j++) {
            int c = tid + j * 128;
            int r = c >> 2, colh = (c & 3) * 8;
            async_copy_b128(qsBase + c * 16, qkv,
                (unsigned)((((b * N_ + qt * 64 + r) * (3 * D_)) + hh * HD_ + colh) * 2));
        }
    }
    issue_ktile(0, 0);
    wait_async<2>();      // Q landed (in-order); K tile 0 may still be in flight
    __syncthreads();
    AFrag qa = load_frag_rowmajor(&Qs[0][0], 32, wave * 16 + am, 0, lane);

    float mrow[8], lrow[8];
    v8f oacc[2];
#pragma unroll
    for (int i = 0; i < 8; i++) { mrow[i] = -__builtin_inff(); lrow[i] = 0.f; }
    oacc[0] = (v8f){0.f,0.f,0.f,0.f,0.f,0.f,0.f,0.f};
    oacc[1] = oacc[0];

#pragma unroll 2
    for (int kt = 0; kt < N_ / 64; kt++) {
        int kb0 = kt * 64;
        int cur = kt & 1;
        __syncthreads();   // prior iteration's Vt / Ks[cur^1] reads are done
        if (kt + 1 < N_ / 64) issue_ktile(cur ^ 1, kb0 + 64);

        // gated + transposed V tile (VALU path, overlaps the async K copies)
        for (int i = tid; i < 64 * 32; i += 128) {
            int r = i >> 5, d = i & 31;
            int key = kb0 + r;
            size_t rowoff = ((size_t)b * N_ + key) * (3 * D_);
            float gte = 1.f;
            if (key < NCTX_)
                gte = 1.f / (1.f + __expf(-tsc * ctx_trust[b * NCTX_ + key]));
            Vt[d][r] = (_Float16)((float)qkv[rowoff + 2 * D_ + hh * HD_ + d] * gte);
        }
        if (tid < 64) {
            int key = kb0 + tid;
            cbias[tid] = (key < NCTX_)
                ? (-alpha * __logf(fmaxf(ctx_ppr[b * NCTX_ + key], 1e-8f)))
                : 0.f;
        }
        if (kt + 1 < N_ / 64) wait_async<2>(); else wait_async<0>();
        __syncthreads();

        // scores: 16 queries x 64 keys  (4 WMMA, K=hd=32 in one shot)
        v8f s[4];
#pragma unroll
        for (int t = 0; t < 4; t++) {
            AFrag kf = load_frag_rowmajor(&Ks[cur][0][0], 32, t * 16 + am, 0, lane);
            v8f c = (v8f){0.f,0.f,0.f,0.f,0.f,0.f,0.f,0.f};
            s[t] = __builtin_amdgcn_wmma_f32_16x16x32_f16(
                false, qa.v, false, kf.v, (short)0, c, false, false);
        }
        // scale + ppr bias + mask (C layout: M = i + 8*(lane>=16), N = lane&15)
#pragma unroll
        for (int t = 0; t < 4; t++) {
            int key = kb0 + t * 16 + (lane & 15);
            float cb = cbias[t * 16 + (lane & 15)];
#pragma unroll
            for (int i = 0; i < 8; i++) {
                int qrow = qt * 64 + wave * 16 + mloc + i;
                bool masked = (key >= NCTX_) && (qrow != key);
                s[t][i] = masked ? -__builtin_inff() : (s[t][i] * scale + cb);
            }
        }
        // online softmax row statistics (reduce across 16-lane halves)
#pragma unroll
        for (int i = 0; i < 8; i++) {
            float tm = fmaxf(fmaxf(s[0][i], s[1][i]), fmaxf(s[2][i], s[3][i]));
#pragma unroll
            for (int off = 1; off < 16; off <<= 1) tm = fmaxf(tm, __shfl_xor(tm, off, 32));
            float mnew = fmaxf(mrow[i], tm);
            float sc = (mrow[i] == -__builtin_inff()) ? 0.f : __expf(mrow[i] - mnew);
            mrow[i] = mnew;
#pragma unroll
            for (int t = 0; t < 4; t++) {
                float p = (mnew == -__builtin_inff()) ? 0.f : __expf(s[t][i] - mnew);
                s[t][i] = p;
            }
            float rs = s[0][i] + s[1][i] + s[2][i] + s[3][i];
#pragma unroll
            for (int off = 1; off < 16; off <<= 1) rs += __shfl_xor(rs, off, 32);
            lrow[i] = lrow[i] * sc + rs;
            oacc[0][i] *= sc;
            oacc[1][i] *= sc;
        }
        // P (C layout) -> LDS row-major so it can be re-read as an A fragment
#pragma unroll
        for (int t = 0; t < 4; t++)
#pragma unroll
            for (int i = 0; i < 8; i++)
                Ps[wave][mloc + i][t * 16 + (lane & 15)] = (_Float16)s[t][i];
        // same-wave LDS RAW: DS ops are in-order within a wave

        // O += P(16x64) @ V(64x32): 2 K chunks x 2 hd tiles
#pragma unroll
        for (int kc = 0; kc < 2; kc++) {
            AFrag pa = load_frag_rowmajor(&Ps[wave][0][0], 64, am, kc * 32, lane);
#pragma unroll
            for (int nt = 0; nt < 2; nt++) {
                AFrag vb = load_frag_rowmajor(&Vt[0][0], 64, nt * 16 + am, kc * 32, lane);
                oacc[nt] = __builtin_amdgcn_wmma_f32_16x16x32_f16(
                    false, pa.v, false, vb.v, (short)0, oacc[nt], false, false);
            }
        }
    }

    // normalize and write o[b, q, h*32 + d]
#pragma unroll
    for (int nt = 0; nt < 2; nt++)
#pragma unroll
        for (int i = 0; i < 8; i++) {
            int qrow = qt * 64 + wave * 16 + mloc + i;
            int d = nt * 16 + (lane & 15);
            float val = oacc[nt][i] / fmaxf(lrow[i], 1e-20f);
            o[((size_t)b * N_ + qrow) * D_ + hh * HD_ + d] = (_Float16)val;
        }
}

// ---------------------------------------------------------------------------
// Orchestration
// ---------------------------------------------------------------------------
extern "C" void kernel_launch(void* const* d_in, const int* in_sizes, int n_in,
                              void* d_out, int out_size, void* d_ws, size_t ws_size,
                              hipStream_t stream) {
    (void)in_sizes; (void)n_in; (void)out_size; (void)ws_size;

    const float* x_in   = (const float*)d_in[0];
    const float* ppr    = (const float*)d_in[1];
    const float* trust  = (const float*)d_in[2];
    const float* w_qkv  = (const float*)d_in[3];
    const float* b_qkv  = (const float*)d_in[4];
    const float* w_out  = (const float*)d_in[5];
    const float* b_out  = (const float*)d_in[6];
    const float* alpha  = (const float*)d_in[7];
    const float* tscale = (const float*)d_in[8];
    const float* g1     = (const float*)d_in[9];
    const float* be1    = (const float*)d_in[10];
    const float* g2     = (const float*)d_in[11];
    const float* be2    = (const float*)d_in[12];
    const float* w_ff1  = (const float*)d_in[13];
    const float* b_ff1  = (const float*)d_in[14];
    const float* w_ff2  = (const float*)d_in[15];
    const float* b_ff2  = (const float*)d_in[16];

    float* x = (float*)d_out;   // running residual stream (f32 master copy)

    // workspace carve-up (halfs)
    _Float16* ws      = (_Float16*)d_ws;
    _Float16* wqkv16  = ws;                                      // L*768*256
    _Float16* wout16  = wqkv16 + (size_t)L_ * 768 * 256;         // L*256*256
    _Float16* wff1_16 = wout16 + (size_t)L_ * 256 * 256;         // L*512*256
    _Float16* wff2_16 = wff1_16 + (size_t)L_ * 512 * 256;        // L*256*512
    _Float16* h16     = wff2_16 + (size_t)L_ * 256 * 512;        // M*256
    _Float16* qkv16   = h16    + (size_t)M_ * 256;               // M*768
    _Float16* o16     = qkv16  + (size_t)M_ * 768;               // M*256
    _Float16* ff16    = o16    + (size_t)M_ * 256;               // M*512

    hipMemcpyAsync(x, x_in, sizeof(float) * (size_t)M_ * D_,
                   hipMemcpyDeviceToDevice, stream);

    auto cvt = [&](const float* src, _Float16* dst, int n) {
        f32_to_f16_kernel<<<(n + 255) / 256, 256, 0, stream>>>(src, dst, n);
    };
    cvt(w_qkv,  wqkv16,  L_ * 768 * 256);
    cvt(w_out,  wout16,  L_ * 256 * 256);
    cvt(w_ff1,  wff1_16, L_ * 512 * 256);
    cvt(w_ff2,  wff2_16, L_ * 256 * 512);

    for (int l = 0; l < L_; l++) {
        // pre-attention LN
        ln_kernel<<<M_ / 8, 256, 0, stream>>>(x, g1 + l * D_, be1 + l * D_, h16);
        // QKV projection: [4096 x 768] = h16 @ Wqkv^T
        gemm_wmma_kernel<256, 0><<<dim3(M_ / 64, 768 / 128), 128, 0, stream>>>(
            h16, wqkv16 + (size_t)l * 768 * 256, b_qkv + l * 768,
            nullptr, qkv16, nullptr, 768);
        // attention
        attn_kernel<<<dim3(B_ * H_, N_ / 64), 128, 0, stream>>>(
            qkv16, ppr, trust, alpha, tscale, l, o16);
        // out projection + residual into x (f32)
        gemm_wmma_kernel<256, 2><<<dim3(M_ / 64, 256 / 128), 128, 0, stream>>>(
            o16, wout16 + (size_t)l * 256 * 256, b_out + l * 256,
            x, nullptr, x, 256);
        // pre-FF LN
        ln_kernel<<<M_ / 8, 256, 0, stream>>>(x, g2 + l * D_, be2 + l * D_, h16);
        // FF1 + exact gelu
        gemm_wmma_kernel<256, 1><<<dim3(M_ / 64, 512 / 128), 128, 0, stream>>>(
            h16, wff1_16 + (size_t)l * 512 * 256, b_ff1 + l * 512,
            nullptr, ff16, nullptr, 512);
        // FF2 + residual into x (f32)
        gemm_wmma_kernel<512, 2><<<dim3(M_ / 64, 256 / 128), 128, 0, stream>>>(
            ff16, wff2_16 + (size_t)l * 256 * 512, b_ff2 + l * 256,
            x, nullptr, x, 256);
    }
}
